// GQAttention_3307124818452
// MI455X (gfx1250) — matrix operand order repaired and checked by reading, not compiled, not run on hardware
//
#include <hip/hip_runtime.h>
#include <hip/hip_bf16.h>
#include <math.h>

typedef __bf16 bf16;
typedef __attribute__((ext_vector_type(16))) __bf16 v16bf;
typedef __attribute__((ext_vector_type(8)))  float  v8f;
typedef __attribute__((ext_vector_type(4)))  unsigned int v4u;
typedef __attribute__((ext_vector_type(2)))  unsigned int v2u;
typedef __attribute__((ext_vector_type(4)))  float  v4f;
typedef __attribute__((ext_vector_type(4)))  unsigned int u32x4;
typedef __attribute__((ext_vector_type(8)))  int  i32x8;
typedef __attribute__((ext_vector_type(4)))  int  i32x4;

union FragBF { v4u u[2]; v16bf v; };

#define QLEN  1024
#define PLEN  1024
#define HIDN  4096
#define NHQ   32
#define NKVH  4
#define DHEAD 128
#define KVLEN 2048   // PLEN + QLEN

// LDS row stride for the TDM-staged A tile: 128 bf16 data + 8 bf16 pad = 272 bytes
#define A_LDS_STRIDE 136

static __device__ __forceinline__ v8f wmma_bf16(const v16bf& a, const v16bf& b, const v8f& c) {
  return __builtin_amdgcn_wmma_f32_16x16x32_bf16(false, a, false, b, (short)0, c, false, false);
}

// ---------------- TDM: load a 16 x 128 bf16 tile (row stride K elems) to LDS --
// D# per CDNA5 ISA. Pads 4 DWORDs after every 64 DWORDs
// (one 256B row) -> LDS row stride 272B, de-conflicting column-aligned reads.
static __device__ __forceinline__ void tdm_load_tile16x128(
    const bf16* gsrc, unsigned lds_byte_addr, int tensor_d0 /*K*/, int tensor_d1 /*M*/) {
  unsigned long long ga = (unsigned long long)gsrc;
  u32x4 g0;
  g0.x = 1u;                                            // count=1, no gather
  g0.y = lds_byte_addr;                                 // lds_addr
  g0.z = (unsigned)(ga & 0xFFFFFFFFull);                // global_addr[31:0]
  g0.w = (unsigned)((ga >> 32) & 0x1FFFFFFull)          // global_addr[56:32]
       | (2u << 30);                                    // type=2 ("image")
  i32x8 g1;
  g1[0] = (int)((1u << 16)      // data_size = 1 -> 2 bytes
              | (1u << 20)      // pad_enable
              | (5u << 22)      // pad_interval: 64 DWORDs
              | (3u << 25));    // pad_amount:   4 DWORDs
  g1[1] = (int)(((unsigned)tensor_d0 & 0xFFFFu) << 16);                       // tensor_dim0 lo
  g1[2] = (int)((((unsigned)tensor_d0 >> 16) & 0xFFFFu)
              | (((unsigned)tensor_d1 & 0xFFFFu) << 16));                     // dim0 hi | dim1 lo
  g1[3] = (int)((((unsigned)tensor_d1 >> 16) & 0xFFFFu) | (128u << 16));      // dim1 hi | tile_dim0=128
  g1[4] = 16;                                                                 // tile_dim1=16
  g1[5] = tensor_d0;                                                          // tensor_dim0_stride = K
  g1[6] = 0;
  g1[7] = 0;
  i32x4 gz4 = {0, 0, 0, 0};                                                   // 2-D: groups 2/3 unused
  i32x8 gz8 = {0, 0, 0, 0, 0, 0, 0, 0};
  // 6-arg form (clang-23 / therock-10.0 headers): (g0, g1, g2, g3, g4, cpol)
  __builtin_amdgcn_tensor_load_to_lds(g0, g1, gz4, gz4, gz8, 0);
}

static __device__ __forceinline__ unsigned lds_offset_of(const void* p) {
  return (unsigned)(unsigned long long)p;   // flat LDS pointer: offset in [31:0]
}

// ---------------- f32 -> bf16 conversion (4-wide) ----------------
__global__ void cvt_f32_bf16(const float* __restrict__ in, bf16* __restrict__ out, int n4) {
  int i = blockIdx.x * blockDim.x + threadIdx.x;
  if (i >= n4) return;
  v4f x = *(const v4f*)(in + (size_t)i * 4);
  union { bf16 h[4]; v2u u; } r;
  r.h[0] = (bf16)x.x; r.h[1] = (bf16)x.y; r.h[2] = (bf16)x.z; r.h[3] = (bf16)x.w;
  *(v2u*)(out + (size_t)i * 4) = r.u;
}

// ---------------- NT GEMM: C[M,N] = A[M,K] * B[N,K]^T (bf16 in, f32 out) -----
// Block = (32,8): 8 waves share one 16(M)-row A tile staged by the TDM into
// LDS (double-buffered, overlapped with WMMA); each wave owns 64 N columns.
__global__ void __launch_bounds__(256)
gemm_nt_bf16(const bf16* __restrict__ A, const bf16* __restrict__ B,
             float* __restrict__ C, int M, int N, int K) {
  __shared__ __align__(16) bf16 atile[2][16 * A_LDS_STRIDE];
  const int lane = threadIdx.x;
  const int wave = threadIdx.y;
  const int half = lane >> 4;
  const int l16  = lane & 15;
  const int m0 = blockIdx.y * 16;
  const int n0 = (blockIdx.x * 8 + wave) * 64;

  v8f acc[4] = {};

  // prologue: stage first 16x128 A chunk
  if (wave == 0) {
    tdm_load_tile16x128(A + (size_t)m0 * K, lds_offset_of(&atile[0][0]), K, M);
    __builtin_amdgcn_s_wait_tensorcnt(0);
  }
  __syncthreads();

  int buf = 0;
  for (int kc = 0; kc < K; kc += 128) {
    // prefetch next chunk into the other buffer (overlaps with compute below)
    if (wave == 0 && kc + 128 < K)
      tdm_load_tile16x128(A + (size_t)m0 * K + kc + 128,
                          lds_offset_of(&atile[buf ^ 1][0]), K, M);
#pragma unroll
    for (int ks = 0; ks < 128; ks += 32) {
      FragBF fa;
      const bf16* ap = &atile[buf][l16 * A_LDS_STRIDE + ks + half * 8];
      fa.u[0] = *(const v4u*)(ap);
      fa.u[1] = *(const v4u*)(ap + 16);
#pragma unroll
      for (int t = 0; t < 4; ++t) {
        const bf16* brow = B + (size_t)(n0 + t * 16 + l16) * K + kc + ks + half * 8;
        FragBF fb;
        fb.u[0] = *(const v4u*)(brow);
        fb.u[1] = *(const v4u*)(brow + 16);
        acc[t] = wmma_bf16(fa.v, fb.v, acc[t]);
      }
    }
    if (wave == 0) __builtin_amdgcn_s_wait_tensorcnt(0);
    __syncthreads();
    buf ^= 1;
  }
#pragma unroll
  for (int t = 0; t < 4; ++t)
#pragma unroll
    for (int v = 0; v < 8; ++v)
      C[(size_t)(m0 + v + half * 8) * N + n0 + t * 16 + l16] = acc[t][v];
}

// ---------------- RoPE helpers ----------------
static __device__ __forceinline__ float rope_angle(int pos, int d) {
  return (float)pos * __powf(10000.0f, -(float)(2 * (d & 63)) * (1.0f / 128.0f));
}

// q f32 [Q][HID] -> bf16 [NHQ][Q][DHEAD] with RoPE at position PLEN+q
__global__ void rope_q_kernel(const float* __restrict__ qf, bf16* __restrict__ qout) {
  int idx = blockIdx.x * blockDim.x + threadIdx.x;
  if (idx >= NHQ * QLEN * DHEAD) return;
  int d = idx & (DHEAD - 1);
  int q = (idx >> 7) & (QLEN - 1);
  int h = idx >> 17;
  const float* row = qf + (size_t)q * HIDN + h * DHEAD;
  float x = row[d];
  float other = (d < 64) ? -row[d + 64] : row[d - 64];
  float ang = rope_angle(PLEN + q, d);
  qout[((size_t)h * QLEN + q) * DHEAD + d] = (bf16)(x * __cosf(ang) + other * __sinf(ang));
}

// concat(past_k, k_new) -> new_past K (f32, pre-RoPE) and RoPE'd bf16 K [NKVH][KVLEN][DHEAD]
__global__ void prep_k_kernel(const float* __restrict__ knew, const float* __restrict__ pastk,
                              float* __restrict__ out_past_k, bf16* __restrict__ krope) {
  int idx = blockIdx.x * blockDim.x + threadIdx.x;
  if (idx >= NKVH * KVLEN * DHEAD) return;
  int d = idx & (DHEAD - 1);
  int pos = (idx >> 7) & (KVLEN - 1);
  int kvh = idx >> 18;
  float x, other;
  if (pos < PLEN) {
    const float* row = pastk + ((size_t)kvh * PLEN + pos) * DHEAD;
    x = row[d]; other = (d < 64) ? -row[d + 64] : row[d - 64];
  } else {
    const float* row = knew + (size_t)(pos - PLEN) * (NKVH * DHEAD) + kvh * DHEAD;
    x = row[d]; other = (d < 64) ? -row[d + 64] : row[d - 64];
  }
  out_past_k[idx] = x;
  float ang = rope_angle(pos, d);
  krope[idx] = (bf16)(x * __cosf(ang) + other * __sinf(ang));
}

// concat(past_v, v_new) -> new_past V (f32) and transposed bf16 V^T [NKVH][DHEAD][KVLEN]
__global__ void prep_v_kernel(const float* __restrict__ vnew, const float* __restrict__ pastv,
                              float* __restrict__ out_past_v, bf16* __restrict__ vT) {
  int idx = blockIdx.x * blockDim.x + threadIdx.x;
  if (idx >= NKVH * KVLEN * DHEAD) return;
  int d = idx & (DHEAD - 1);
  int pos = (idx >> 7) & (KVLEN - 1);
  int kvh = idx >> 18;
  float x;
  if (pos < PLEN) x = pastv[((size_t)kvh * PLEN + pos) * DHEAD + d];
  else            x = vnew[(size_t)(pos - PLEN) * (NKVH * DHEAD) + kvh * DHEAD + d];
  out_past_v[idx] = x;
  vT[((size_t)kvh * DHEAD + d) * KVLEN + pos] = (bf16)x;
}

// ---------------- Flash attention: one wave per 16-row Q tile ----------------
// grid = (QLEN/16/8, NHQ), block = (32,8)
__global__ void __launch_bounds__(256)
attn_kernel(const bf16* __restrict__ qr, const bf16* __restrict__ kr,
            const bf16* __restrict__ vT, bf16* __restrict__ ctx) {
  __shared__ float plds[8][16][33];
  const int lane = threadIdx.x;
  const int wave = threadIdx.y;
  const int half = lane >> 4;
  const int l16  = lane & 15;
  const int h    = blockIdx.y;
  const int kvh  = h >> 3;                         // N_REP = 8
  const int qt   = blockIdx.x * 8 + wave;
  const int q0   = qt * 16;

  // Q A-fragments (16 rows x 128 d, 4 chunks of 32)
  FragBF fq[4];
  const bf16* qrow = qr + ((size_t)h * QLEN + q0 + l16) * DHEAD + half * 8;
#pragma unroll
  for (int f = 0; f < 4; ++f) {
    fq[f].u[0] = *(const v4u*)(qrow + f * 32);
    fq[f].u[1] = *(const v4u*)(qrow + f * 32 + 16);
  }

  v8f acc[8] = {};
  float m[8], lsum[8];
#pragma unroll
  for (int v = 0; v < 8; ++v) { m[v] = -INFINITY; lsum[v] = 0.0f; }

  const float scale = 0.08838834764831845f;        // 1/sqrt(128)
  const int qpos_base = PLEN + q0 + half * 8;      // row v -> qpos_base + v
  // uniform across the block so __syncthreads is legal; extra tiles are fully masked
  const int kv_end = PLEN + (blockIdx.x * 8 + 7) * 16 + 16;

  for (int kb = 0; kb < kv_end; kb += 32) {
    // ---- scores: two 16x16 tiles over kv = kb..kb+31
    v8f s[2];
#pragma unroll
    for (int t2 = 0; t2 < 2; ++t2) {
      v8f sc = {};
      const bf16* krow = kr + ((size_t)kvh * KVLEN + kb + t2 * 16 + l16) * DHEAD + half * 8;
#pragma unroll
      for (int f = 0; f < 4; ++f) {
        FragBF fk;
        fk.u[0] = *(const v4u*)(krow + f * 32);
        fk.u[1] = *(const v4u*)(krow + f * 32 + 16);
        sc = wmma_bf16(fq[f].v, fk.v, sc);
      }
      s[t2] = sc;
    }
    // ---- scale + causal mask
#pragma unroll
    for (int t2 = 0; t2 < 2; ++t2) {
      int kvp = kb + t2 * 16 + l16;
#pragma unroll
      for (int v = 0; v < 8; ++v) {
        float val = s[t2][v] * scale;
        if (kvp > qpos_base + v) val += -1.0e9f;
        s[t2][v] = val;
      }
    }
    // ---- online softmax (row reductions across 16-lane halves)
    float alpha[8];
#pragma unroll
    for (int v = 0; v < 8; ++v) {
      float mx = fmaxf(s[0][v], s[1][v]);
#pragma unroll
      for (int off = 1; off < 16; off <<= 1) mx = fmaxf(mx, __shfl_xor(mx, off, 16));
      float mn = fmaxf(m[v], mx);
      alpha[v] = __expf(m[v] - mn);
      m[v] = mn;
    }
#pragma unroll
    for (int t2 = 0; t2 < 2; ++t2)
#pragma unroll
      for (int v = 0; v < 8; ++v)
        s[t2][v] = __expf(s[t2][v] - m[v]);
#pragma unroll
    for (int v = 0; v < 8; ++v) {
      float sm = s[0][v] + s[1][v];
#pragma unroll
      for (int off = 1; off < 16; off <<= 1) sm += __shfl_xor(sm, off, 16);
      lsum[v] = lsum[v] * alpha[v] + sm;
    }
#pragma unroll
    for (int t = 0; t < 8; ++t)
#pragma unroll
      for (int v = 0; v < 8; ++v)
        acc[t][v] *= alpha[v];

    // ---- transpose probs C-layout -> A-layout through LDS
    __syncthreads();   // WAR vs previous iteration's reads
#pragma unroll
    for (int t2 = 0; t2 < 2; ++t2)
#pragma unroll
      for (int v = 0; v < 8; ++v)
        plds[wave][v + half * 8][t2 * 16 + l16] = s[t2][v];
    __syncthreads();   // RAW
    v16bf pv;
#pragma unroll
    for (int e = 0; e < 16; ++e) {
      int kk = (e & 7) + ((e >> 3) << 4) + half * 8;   // A-layout K index
      pv[e] = (bf16)plds[wave][l16][kk];
    }
    // ---- ctx += P(16x32) * V(32x128), 8 d-tiles
#pragma unroll
    for (int t = 0; t < 8; ++t) {
      const bf16* vrow = vT + ((size_t)kvh * DHEAD + t * 16 + l16) * KVLEN + kb + half * 8;
      FragBF fv;
      fv.u[0] = *(const v4u*)(vrow);
      fv.u[1] = *(const v4u*)(vrow + 16);
      acc[t] = wmma_bf16(pv, fv.v, acc[t]);
    }
  }

  // ---- normalize and store ctx (bf16, [q][HID] so O-proj reads it like hidden)
#pragma unroll
  for (int v = 0; v < 8; ++v) {
    float inv = 1.0f / lsum[v];
#pragma unroll
    for (int t = 0; t < 8; ++t)
      ctx[(size_t)(q0 + v + half * 8) * HIDN + h * DHEAD + t * 16 + l16] =
          (bf16)(acc[t][v] * inv);
  }
}

// ---------------- host-side orchestration ----------------
extern "C" void kernel_launch(void* const* d_in, const int* in_sizes, int n_in,
                              void* d_out, int out_size, void* d_ws, size_t ws_size,
                              hipStream_t stream) {
  const float* hs   = (const float*)d_in[0];
  const float* past = (const float*)d_in[1];   // [1,2,NKVH,PLEN,DHEAD]
  const float* Wq   = (const float*)d_in[3];
  const float* Wk   = (const float*)d_in[4];
  const float* Wv   = (const float*)d_in[5];
  const float* Wo   = (const float*)d_in[6];
  float* out = (float*)d_out;

  char* ws = (char*)d_ws;
  size_t off = 0;
  auto take = [&](size_t bytes) -> void* {
    void* p = ws + off;
    off += (bytes + 255) & ~(size_t)255;
    return p;
  };
  bf16* hs_bf  = (bf16*)take((size_t)QLEN * HIDN * 2);
  bf16* wq_bf  = (bf16*)take((size_t)HIDN * HIDN * 2);
  bf16* wk_bf  = (bf16*)take((size_t)NKVH * DHEAD * HIDN * 2);
  bf16* wv_bf  = (bf16*)take((size_t)NKVH * DHEAD * HIDN * 2);
  bf16* wo_bf  = (bf16*)take((size_t)HIDN * HIDN * 2);
  float* qf32  = (float*)take((size_t)QLEN * HIDN * 4);
  float* kf32  = (float*)take((size_t)QLEN * NKVH * DHEAD * 4);
  float* vf32  = (float*)take((size_t)QLEN * NKVH * DHEAD * 4);
  bf16* q_rope = (bf16*)take((size_t)NHQ * QLEN * DHEAD * 2);
  bf16* k_rope = (bf16*)take((size_t)NKVH * KVLEN * DHEAD * 2);
  bf16* vT_bf  = (bf16*)take((size_t)NKVH * DHEAD * KVLEN * 2);
  bf16* ctx_bf = (bf16*)take((size_t)QLEN * HIDN * 2);

  // 1) f32 -> bf16 conversions
  auto cvt = [&](const float* src, bf16* dst, size_t n) {
    int n4 = (int)(n / 4);
    cvt_f32_bf16<<<(n4 + 255) / 256, 256, 0, stream>>>(src, dst, n4);
  };
  cvt(hs, hs_bf, (size_t)QLEN * HIDN);
  cvt(Wq, wq_bf, (size_t)HIDN * HIDN);
  cvt(Wk, wk_bf, (size_t)NKVH * DHEAD * HIDN);
  cvt(Wv, wv_bf, (size_t)NKVH * DHEAD * HIDN);
  cvt(Wo, wo_bf, (size_t)HIDN * HIDN);

  // 2) projections: C[M,N] = hs * W^T
  dim3 blk(32, 8);
  gemm_nt_bf16<<<dim3(HIDN / 512, QLEN / 16), blk, 0, stream>>>(hs_bf, wq_bf, qf32, QLEN, HIDN, HIDN);
  gemm_nt_bf16<<<dim3((NKVH * DHEAD) / 512, QLEN / 16), blk, 0, stream>>>(hs_bf, wk_bf, kf32, QLEN, NKVH * DHEAD, HIDN);
  gemm_nt_bf16<<<dim3((NKVH * DHEAD) / 512, QLEN / 16), blk, 0, stream>>>(hs_bf, wv_bf, vf32, QLEN, NKVH * DHEAD, HIDN);

  // 3) RoPE + KV-cache concat (new_past is pre-RoPE, matching the reference)
  const float* past_k = past;
  const float* past_v = past + (size_t)NKVH * PLEN * DHEAD;
  float* out_attn   = out;
  float* out_past_k = out + (size_t)QLEN * HIDN;
  float* out_past_v = out_past_k + (size_t)NKVH * KVLEN * DHEAD;

  rope_q_kernel<<<(NHQ * QLEN * DHEAD) / 256, 256, 0, stream>>>(qf32, q_rope);
  prep_k_kernel<<<(NKVH * KVLEN * DHEAD) / 256, 256, 0, stream>>>(kf32, past_k, out_past_k, k_rope);
  prep_v_kernel<<<(NKVH * KVLEN * DHEAD) / 256, 256, 0, stream>>>(vf32, past_v, out_past_v, vT_bf);

  // 4) flash attention -> ctx (bf16)
  attn_kernel<<<dim3(QLEN / 16 / 8, NHQ), blk, 0, stream>>>(q_rope, k_rope, vT_bf, ctx_bf);

  // 5) output projection -> f32 attn_output
  gemm_nt_bf16<<<dim3(HIDN / 512, QLEN / 16), blk, 0, stream>>>(ctx_bf, wo_bf, out_attn, QLEN, HIDN, HIDN);
}
